// Net_56272661512546
// MI455X (gfx1250) — matrix operand order repaired and checked by reading, not compiled, not run on hardware
//
#include <hip/hip_runtime.h>
#include <hip/hip_bf16.h>

// ---------------------------------------------------------------------------
// Problem constants (from the reference)
// ---------------------------------------------------------------------------
#define BB   128   // batch
#define CIN  3
#define HW0  80    // input H=W
#define C1   20
#define HW1  78    // after conv1
#define C2   30
#define HW2  76    // after conv2
#define KK   32    // keypoints / conv3 out channels
#define HW3  37    // after strided conv3
#define GRD  80    // softmax grid

typedef __attribute__((ext_vector_type(16))) _Float16 v16h;
typedef __attribute__((ext_vector_type(8)))  _Float16 v8h;
typedef __attribute__((ext_vector_type(8)))  float    v8f;

// ---------------------------------------------------------------------------
// Block reductions (256 threads)
// ---------------------------------------------------------------------------
__device__ __forceinline__ float blockSum(float v, float* red) {
  int tid = threadIdx.x;
  red[tid] = v; __syncthreads();
  for (int off = 128; off > 0; off >>= 1) {
    if (tid < off) red[tid] += red[tid + off];
    __syncthreads();
  }
  float r = red[0]; __syncthreads();
  return r;
}
__device__ __forceinline__ float blockMax(float v, float* red) {
  int tid = threadIdx.x;
  red[tid] = v; __syncthreads();
  for (int off = 128; off > 0; off >>= 1) {
    if (tid < off) red[tid] = fmaxf(red[tid], red[tid + off]);
    __syncthreads();
  }
  float r = red[0]; __syncthreads();
  return r;
}

// ---------------------------------------------------------------------------
// Kernel 1: conv1 (3->20, 3x3, VALID) direct fp32, input bias -0.33, relu,
// store f16 activations for the WMMA layers.
// ---------------------------------------------------------------------------
__global__ void k_conv1(const float* __restrict__ x, const float* __restrict__ w1,
                        const float* __restrict__ b1, _Float16* __restrict__ x2) {
  int idx = blockIdx.x * 256 + threadIdx.x;
  const int total = BB * C1 * HW1 * HW1;
  if (idx >= total) return;
  int ox = idx % HW1; int t = idx / HW1;
  int oy = t % HW1;   t /= HW1;
  int oc = t % C1;    int b = t / C1;
  float acc = b1[oc];
  #pragma unroll
  for (int ic = 0; ic < CIN; ++ic)
    #pragma unroll
    for (int ky = 0; ky < 3; ++ky)
      #pragma unroll
      for (int kx = 0; kx < 3; ++kx) {
        float v = x[((size_t)(b * CIN + ic) * HW0 + (oy + ky)) * HW0 + (ox + kx)] - 0.33f;
        acc += v * w1[((oc * CIN + ic) * 3 + ky) * 3 + kx];
      }
  x2[idx] = (_Float16)fmaxf(acc, 0.f);
}

// ---------------------------------------------------------------------------
// Kernel 2: pack conv weights into padded im2col matrices, f16.
//   W2: [32][192]  (n<30 valid, k<180 valid, k = ic*9+ky*3+kx)
//   W3: [32][288]  (k<270 valid)
// Row-major with K contiguous -> B-fragment loads are contiguous 32B.
// ---------------------------------------------------------------------------
__global__ void k_prep_w(const float* __restrict__ w2, const float* __restrict__ w3,
                         _Float16* __restrict__ W2, _Float16* __restrict__ W3) {
  int i = blockIdx.x * 256 + threadIdx.x;
  if (i < 32 * 192) {
    int n = i / 192, k = i % 192;
    float v = 0.f;
    if (n < C2 && k < C1 * 9) {
      int ic = k / 9, r = k % 9;
      v = w2[((n * C1 + ic) * 3 + r / 3) * 3 + (r % 3)];
    }
    W2[i] = (_Float16)v;
  }
  if (i < 32 * 288) {
    int n = i / 288, k = i % 288;
    float v = 0.f;
    if (k < C2 * 9) {
      int ic = k / 9, r = k % 9;
      v = w3[((n * C2 + ic) * 3 + r / 3) * 3 + (r % 3)];
    }
    W3[i] = (_Float16)v;
  }
}

// ---------------------------------------------------------------------------
// Kernel 3: implicit-GEMM conv via WMMA f16 (16x16x32), f32 accumulate.
// Block = 128 threads = 4 wave32; each wave computes a 16-pixel x 32-channel
// tile; 64 output pixels per block. A (im2col) and B (weights) staged in LDS.
//
// Fragment layouts used here (16-bit, wave32; contiguous K, not the ISA table order):
//   A 16x32 : lane m=lane&15 is row M; khalf=lane>>4; elems 0..15 -> K=khalf*16+e
//             (one contiguous 16-half run, matching B)
//   B 32x16 : lane n=lane&15 is col N; lanes 0-15 hold K=0..15, lanes 16-31
//             hold K=16..31 (one contiguous 16-half run)
//   C 16x16 : N=lane&15, M=(lane>>4)*8 + r for v8f element r
// ---------------------------------------------------------------------------
template<int CINV, int HIN, int COUTV, int HOUT, int KPAD, int STRIDE,
         bool RELU, bool OUT_F16>
__global__ __launch_bounds__(128)
void k_conv_wmma(const _Float16* __restrict__ xin, const _Float16* __restrict__ Wmat,
                 const float* __restrict__ bias, void* __restrict__ out, int pixTiles) {
  constexpr int KV     = CINV * 9;       // valid K
  constexpr int MPIX   = HOUT * HOUT;    // output pixels per image
  constexpr int KSTEPS = KPAD / 32;

  __shared__ __align__(32) _Float16 Atile[64 * KPAD];
  __shared__ __align__(32) _Float16 Wlds [32 * KPAD];

  const int b       = blockIdx.x / pixTiles;
  const int tile    = blockIdx.x % pixTiles;
  const int pixbase = tile * 64;
  const int tid     = threadIdx.x;

  // speculative L2 warm-up for the NEXT pixel tile's input rows
  // (__builtin_prefetch -> global_prefetch_b8; im2col re-reads each input
  //  ~9x, all of it L2-resident: x2 is 31 MB << 192 MB L2)
  {
    int npix = pixbase + 64 + (tid & 63);
    if (npix < MPIX) {
      int oy = npix / HOUT, ox = npix % HOUT;
      __builtin_prefetch(&xin[((size_t)(b * CINV) * HIN + oy * STRIDE) * HIN + ox * STRIDE], 0, 1);
    }
  }

  // stage weights as 16-byte chunks (global_load_b128 + ds_store_b128)
  {
    const v8h* wsrc = (const v8h*)Wmat;
    v8h* wdst = (v8h*)Wlds;
    for (int i = tid; i < (32 * KPAD) / 8; i += 128) wdst[i] = wsrc[i];
  }

  // im2col into LDS (zero-pad K tail and out-of-range pixels)
  for (int i = tid; i < 64 * KPAD; i += 128) {
    int p = i / KPAD, k = i % KPAD;
    _Float16 v = (_Float16)0.f;
    int pix = pixbase + p;
    if (pix < MPIX && k < KV) {
      int oy = pix / HOUT, ox = pix % HOUT;
      int ic = k / 9, r = k % 9, ky = r / 3, kx = r % 3;
      v = xin[((size_t)(b * CINV + ic) * HIN + (oy * STRIDE + ky)) * HIN + (ox * STRIDE + kx)];
    }
    Atile[i] = v;
  }
  __syncthreads();

  const int lane  = tid & 31;
  const int wave  = tid >> 5;
  const int m     = lane & 15;
  const int khalf = lane >> 4;

  v8f acc0 = {0.f,0.f,0.f,0.f,0.f,0.f,0.f,0.f};
  v8f acc1 = {0.f,0.f,0.f,0.f,0.f,0.f,0.f,0.f};

  const _Float16* arow  = &Atile[(wave * 16 + m) * KPAD];
  const _Float16* brow0 = &Wlds [(lane & 15) * KPAD];
  const _Float16* brow1 = &Wlds [(16 + (lane & 15)) * KPAD];

  #pragma unroll
  for (int s = 0; s < KSTEPS; ++s) {
    v8h alo = *(const v8h*)(arow + s * 32 + khalf * 16);
    v8h ahi = *(const v8h*)(arow + s * 32 + khalf * 16 + 8);
    v16h a;
    #pragma unroll
    for (int e = 0; e < 8; ++e) { a[e] = alo[e]; a[8 + e] = ahi[e]; }
    v16h bf0 = *(const v16h*)(brow0 + s * 32 + khalf * 16);
    v16h bf1 = *(const v16h*)(brow1 + s * 32 + khalf * 16);
    acc0 = __builtin_amdgcn_wmma_f32_16x16x32_f16(false, a, false, bf0, (short)0, acc0, false, false);
    acc1 = __builtin_amdgcn_wmma_f32_16x16x32_f16(false, a, false, bf1, (short)0, acc1, false, false);
  }

  // epilogue: bias (+relu), scatter to NCHW output
  const int n = lane & 15;
  #pragma unroll
  for (int r = 0; r < 8; ++r) {
    int M   = khalf * 8 + r;
    int pix = pixbase + wave * 16 + M;
    if (pix < MPIX) {
      int py = pix / HOUT, px = pix % HOUT;
      if (n < COUTV) {
        float v = acc0[r] + bias[n];
        if (RELU) v = fmaxf(v, 0.f);
        size_t o = ((size_t)(b * COUTV + n) * HOUT + py) * HOUT + px;
        if constexpr (OUT_F16) ((_Float16*)out)[o] = (_Float16)v;
        else                   ((float*)out)[o] = v;
      }
      int ch = 16 + n;
      if (ch < COUTV) {
        float v = acc1[r] + bias[ch];
        if (RELU) v = fmaxf(v, 0.f);
        size_t o = ((size_t)(b * COUTV + ch) * HOUT + py) * HOUT + px;
        if constexpr (OUT_F16) ((_Float16*)out)[o] = (_Float16)v;
        else                   ((float*)out)[o] = v;
      }
    }
  }
}

// ---------------------------------------------------------------------------
// Kernel 4: per-(b,k) bilinear 37->80 upsample (on the fly from LDS) +
// spatial softmax + soft-argmax. One block per (b,k).
// jax.image.resize bilinear: src = (dst+0.5)*(37/80)-0.5, edge clamped
// (antialias is a no-op for upsampling).
// ---------------------------------------------------------------------------
__global__ __launch_bounds__(256)
void k_softmax(const float* __restrict__ h, float* __restrict__ s_out,
               float* __restrict__ kp_out, float* __restrict__ xv,
               float* __restrict__ yv) {
  __shared__ float hb[HW3 * HW3];
  __shared__ float red[256];
  const int bk = blockIdx.x, tid = threadIdx.x;
  const float* hp = h + (size_t)bk * (HW3 * HW3);
  for (int i = tid; i < HW3 * HW3; i += 256) hb[i] = hp[i];
  __syncthreads();

  auto bil = [&](int i, int j) -> float {
    const float sc = (float)HW3 / (float)GRD;
    float sy = (i + 0.5f) * sc - 0.5f;
    float sx = (j + 0.5f) * sc - 0.5f;
    int y0 = (int)floorf(sy); float fy = sy - (float)y0;
    int x0 = (int)floorf(sx); float fx = sx - (float)x0;
    int y0c = min(max(y0, 0), HW3 - 1), y1c = min(max(y0 + 1, 0), HW3 - 1);
    int x0c = min(max(x0, 0), HW3 - 1), x1c = min(max(x0 + 1, 0), HW3 - 1);
    float v00 = hb[y0c * HW3 + x0c], v01 = hb[y0c * HW3 + x1c];
    float v10 = hb[y1c * HW3 + x0c], v11 = hb[y1c * HW3 + x1c];
    return (1.f - fy) * ((1.f - fx) * v00 + fx * v01) +
           fy * ((1.f - fx) * v10 + fx * v11);
  };

  // pass 1: max
  float mx = -1e30f;
  for (int t = tid; t < GRD * GRD; t += 256) mx = fmaxf(mx, bil(t / GRD, t % GRD));
  mx = blockMax(mx, red);

  // pass 2: sum of exp + coord moments
  float se = 0.f, si = 0.f, sj = 0.f;
  for (int t = tid; t < GRD * GRD; t += 256) {
    int i = t / GRD, j = t % GRD;
    float e = expf(bil(i, j) - mx);
    se += e; si += e * (float)i; sj += e * (float)j;
  }
  se = blockSum(se, red);
  si = blockSum(si, red);
  sj = blockSum(sj, red);
  float inv = 1.f / se;
  float x_v = si * inv;   // row coord
  float y_v = sj * inv;   // col coord

  // pass 3: write normalized softmax map
  float* sp = s_out + (size_t)bk * (GRD * GRD);
  for (int t = tid; t < GRD * GRD; t += 256)
    sp[t] = expf(bil(t / GRD, t % GRD) - mx) * inv;

  if (tid == 0) {
    kp_out[bk * 2 + 0] = x_v;
    kp_out[bk * 2 + 1] = y_v;
    xv[bk] = x_v; yv[bk] = y_v;
  }
}

// ---------------------------------------------------------------------------
// Kernel 5: frame-difference silhouette  (sum_c |prev-first| > 0 -> 1.0)
// ---------------------------------------------------------------------------
__global__ void k_change(const float* __restrict__ first, const float* __restrict__ prev,
                         float* __restrict__ chg) {
  int idx = blockIdx.x * 256 + threadIdx.x;
  const int total = BB * GRD * GRD;
  if (idx >= total) return;
  int b = idx / (GRD * GRD), p = idx % (GRD * GRD);
  const float* f = first + (size_t)b * CIN * GRD * GRD + p;
  const float* q = prev  + (size_t)b * CIN * GRD * GRD + p;
  float v = fabsf(q[0] - f[0]) + fabsf(q[GRD * GRD] - f[GRD * GRD]) +
            fabsf(q[2 * GRD * GRD] - f[2 * GRD * GRD]);
  chg[idx] = (v > 0.f) ? 1.f : 0.f;
}

// Kernel 5b: per-batch silhouette mass (for maskb)
__global__ __launch_bounds__(256)
void k_sumchg(const float* __restrict__ chg, float* __restrict__ schg) {
  __shared__ float red[256];
  int b = blockIdx.x, tid = threadIdx.x;
  const float* cp = chg + (size_t)b * GRD * GRD;
  float s = 0.f;
  for (int t = tid; t < GRD * GRD; t += 256) s += cp[t];
  s = blockSum(s, red);
  if (tid == 0) schg[b] = s;
}

// ---------------------------------------------------------------------------
// Kernel 6: per-(b,k) reductions: inner = <s, chg>, varsum = <s, d2map>
// ---------------------------------------------------------------------------
__global__ __launch_bounds__(256)
void k_perbk(const float* __restrict__ s, const float* __restrict__ chg,
             const float* __restrict__ xv, const float* __restrict__ yv,
             float* __restrict__ inner, float* __restrict__ var) {
  __shared__ float red[256];
  int bk = blockIdx.x, tid = threadIdx.x;
  int b = bk >> 5;
  const float* sp = s   + (size_t)bk * GRD * GRD;
  const float* cp = chg + (size_t)b  * GRD * GRD;
  float x_v = xv[bk], y_v = yv[bk];
  float a = 0.f, v = 0.f;
  for (int t = tid; t < GRD * GRD; t += 256) {
    float sv = sp[t];
    a += sv * cp[t];
    float di = (float)(t / GRD) - x_v;
    float dj = (float)(t % GRD) - y_v;
    v += sv * (di * di + dj * dj);
  }
  a = blockSum(a, red);
  v = blockSum(v, red);
  if (tid == 0) { inner[bk] = a; var[bk] = v; }
}

// ---------------------------------------------------------------------------
// Kernel 7: finalize losses [variety, consistency, variance]
// ---------------------------------------------------------------------------
__global__ __launch_bounds__(256)
void k_losses(const float* __restrict__ xv, const float* __restrict__ yv,
              const float* __restrict__ inner, const float* __restrict__ var,
              const float* __restrict__ schg, float* __restrict__ losses) {
  __shared__ float red[256];
  int tid = threadIdx.x;

  // variance loss: mean over (b,k)
  float v = 0.f;
  for (int t = tid; t < BB * KK; t += 256) v += var[t];
  v = blockSum(v, red);
  float var_loss = v / (float)(BB * KK);

  // consistency
  float cv = 0.f;
  for (int t = tid; t < BB * KK; t += 256) {
    int b = t >> 5;
    if (schg[b] > 0.f) cv += -logf(1e-7f + inner[t]);
  }
  cv = blockSum(cv, red);
  float cm = 0.f;
  if (tid < BB) cm = (schg[tid] > 0.f) ? 1.f : 0.f;
  cm = blockSum(cm, red);
  float cons = cv / (cm * (float)KK);

  // variety: sum_{b,i!=j} relu(100 - ||kp_i - kp_j||^2) / (B*K^2)
  float vr = 0.f;
  for (int t = tid; t < BB * KK * KK; t += 256) {
    int b = t / (KK * KK), r = t % (KK * KK);
    int i = r / KK, j = r % KK;
    if (i != j) {
      float dx = xv[b * KK + i] - xv[b * KK + j];
      float dy = yv[b * KK + i] - yv[b * KK + j];
      vr += fmaxf(100.f - (dx * dx + dy * dy), 0.f);
    }
  }
  vr = blockSum(vr, red);
  float variety = vr / (float)(BB * KK * KK);

  if (tid == 0) {
    losses[0] = variety;
    losses[1] = cons;
    losses[2] = var_loss;
  }
}

// ---------------------------------------------------------------------------
// Host-side orchestration
// ---------------------------------------------------------------------------
extern "C" void kernel_launch(void* const* d_in, const int* in_sizes, int n_in,
                              void* d_out, int out_size, void* d_ws, size_t ws_size,
                              hipStream_t stream) {
  const float* first = (const float*)d_in[0];
  const float* prev  = (const float*)d_in[1];
  const float* w1    = (const float*)d_in[2];
  const float* b1    = (const float*)d_in[3];
  const float* w2    = (const float*)d_in[4];
  const float* b2    = (const float*)d_in[5];
  const float* w3    = (const float*)d_in[6];
  const float* b3    = (const float*)d_in[7];

  char* ws = (char*)d_ws;
  size_t o = 0;
  auto alloc = [&](size_t bytes) { size_t r = o; o = (o + bytes + 255) & ~(size_t)255; return r; };

  _Float16* x2   = (_Float16*)(ws + alloc(sizeof(_Float16) * (size_t)BB * C1 * HW1 * HW1));
  _Float16* x3   = (_Float16*)(ws + alloc(sizeof(_Float16) * (size_t)BB * C2 * HW2 * HW2));
  float*    hbuf = (float*)   (ws + alloc(sizeof(float)    * (size_t)BB * KK * HW3 * HW3));
  _Float16* W2   = (_Float16*)(ws + alloc(sizeof(_Float16) * 32 * 192));
  _Float16* W3   = (_Float16*)(ws + alloc(sizeof(_Float16) * 32 * 288));
  float*    xv   = (float*)(ws + alloc(sizeof(float) * BB * KK));
  float*    yv   = (float*)(ws + alloc(sizeof(float) * BB * KK));
  float*    innr = (float*)(ws + alloc(sizeof(float) * BB * KK));
  float*    varb = (float*)(ws + alloc(sizeof(float) * BB * KK));
  float*    schg = (float*)(ws + alloc(sizeof(float) * BB));

  // d_out layout: kp [B,K,2] | s [B,K,80,80] | img_change [B,80,80] | losses [3]
  float* out_kp  = (float*)d_out;
  float* out_s   = out_kp + (size_t)BB * KK * 2;
  float* out_chg = out_s  + (size_t)BB * KK * GRD * GRD;
  float* out_ls  = out_chg + (size_t)BB * GRD * GRD;

  // 1) conv1 direct
  {
    int total = BB * C1 * HW1 * HW1;
    k_conv1<<<(total + 255) / 256, 256, 0, stream>>>(first, w1, b1, x2);
  }
  // 2) weight packing
  k_prep_w<<<(32 * 288 + 255) / 256, 256, 0, stream>>>(w2, w3, W2, W3);

  // 3) conv2 via WMMA: M=76*76 per image, K=180->192, N=30->32
  {
    const int pixTiles = (HW2 * HW2 + 63) / 64;  // 91
    k_conv_wmma<C1, HW1, C2, HW2, 192, 1, true, true>
        <<<BB * pixTiles, 128, 0, stream>>>(x2, W2, b2, (void*)x3, pixTiles);
  }
  // 4) conv3 (stride 2) via WMMA: M=37*37, K=270->288, N=32
  {
    const int pixTiles = (HW3 * HW3 + 63) / 64;  // 22
    k_conv_wmma<C2, HW2, KK, HW3, 288, 2, false, false>
        <<<BB * pixTiles, 128, 0, stream>>>(x3, W3, b3, (void*)hbuf, pixTiles);
  }
  // 5) bilinear + softmax + soft-argmax
  k_softmax<<<BB * KK, 256, 0, stream>>>(hbuf, out_s, out_kp, xv, yv);

  // 6) silhouette + per-batch mass
  {
    int total = BB * GRD * GRD;
    k_change<<<(total + 255) / 256, 256, 0, stream>>>(first, prev, out_chg);
  }
  k_sumchg<<<BB, 256, 0, stream>>>(out_chg, schg);

  // 7) per-(b,k) reductions
  k_perbk<<<BB * KK, 256, 0, stream>>>(out_s, out_chg, xv, yv, innr, varb);

  // 8) losses
  k_losses<<<1, 256, 0, stream>>>(xv, yv, innr, varb, schg, out_ls);
}
